// GNN_65360812310791
// MI455X (gfx1250) — matrix operand order repaired and checked by reading, not compiled, not run on hardware
//
#include <hip/hip_runtime.h>
#include <hip/hip_bf16.h>

// ---------------------------------------------------------------- constants
#define B_      1024
#define O_      20
#define V_      400
#define E_      840
#define N_      421                 // 1 + O + V
#define H_      128
#define HEADS_  4
#define D_      32
#define LAYERS_ 2
#define M_      (B_ * N_)           // 431104 nodes  (divisible by 16)
#define NE_     (E_ + N_)           // 1261 edges per graph (incl. self loops)
#define NEG_SLOPE_ 0.2f
#define FMIN_   (-3.402823466e38f)

typedef __attribute__((ext_vector_type(16))) _Float16 v16h;
typedef __attribute__((ext_vector_type(8)))  _Float16 v8h;
typedef __attribute__((ext_vector_type(8)))  float    v8f;

union AFrag { v16h v; v8h h[2]; };

// -------- fused-attention LDS layout (bytes), total 272760 < 320K per WGP
#define SM_AGG   0                       // float[N*H]        215552 B
#define SM_ALS   215552                  // float[N*HEADS]      6736 B
#define SM_ALD   222288                  // float[N*HEADS]      6736 B
#define SM_EMX   229024                  // float[N*HEADS]      6736 B
#define SM_DEN   235760                  // float[N*HEADS]      6736 B
#define SM_EX    242496                  // float[NE*HEADS]    20176 B
#define SM_SL    262672                  // int[NE]             5044 B
#define SM_DL    267716                  // int[NE]             5044 B
#define SM_TOTAL 272760

// ---------------------------------------------------------------- embedding
// x[m, c] = relu(feat(m) . We_type[:, c] + be_type[c]); also emit f16 copy.
__global__ __launch_bounds__(256) void k_embed(
    const float* __restrict__ head, const float* __restrict__ obj,
    const float* __restrict__ val,
    const float* __restrict__ We_head, const float* __restrict__ be_head,
    const float* __restrict__ We_obj,  const float* __restrict__ be_obj,
    const float* __restrict__ We_val,  const float* __restrict__ be_val,
    float* __restrict__ x32, _Float16* __restrict__ x16)
{
    int i = blockIdx.x * 256 + threadIdx.x;      // i in [0, M*128)
    int m = i >> 7;
    int c = i & 127;
    int b = m / N_;
    int r = m - b * N_;
    float acc;
    if (r == 0) {
        const float* f = head + b * 2;
        acc = be_head[c] + f[0] * We_head[c] + f[1] * We_head[H_ + c];
    } else if (r <= O_) {
        const float* f = obj + (b * O_ + (r - 1)) * 2;
        acc = be_obj[c] + f[0] * We_obj[c] + f[1] * We_obj[H_ + c];
    } else {
        const float* f = val + ((size_t)b * V_ + (r - 1 - O_)) * 5;
        acc = be_val[c];
        #pragma unroll
        for (int q = 0; q < 5; ++q) acc += f[q] * We_val[q * H_ + c];
    }
    acc = acc > 0.f ? acc : 0.f;
    x32[i] = acc;
    x16[i] = (_Float16)acc;
}

// ---------------------------------------------------------------- W packing
// Pack W_gat[l] (128x128 fp32, k-major) into WMMA f16 B fragments (contiguous k order, same as A):
// fragment (t,kk): lane l holds column n = t*16 + (l&15), halfs j=0..15 hold
// K = kk*32 + (l>>4)*16 + j.  One contiguous v16h per lane per fragment.
__global__ __launch_bounds__(256) void k_packW(
    const float* __restrict__ Wg, _Float16* __restrict__ wpack)
{
    int tid = blockIdx.x * 256 + threadIdx.x;    // [0, LAYERS*16384)
    int layer = tid >> 14;
    int r = tid & 16383;
    int j  = r & 15;
    int ln = (r >> 4) & 31;
    int kk = (r >> 9) & 3;
    int t  = r >> 11;
    int n  = t * 16 + (ln & 15);
    int k  = kk * 32 + (ln >> 4) * 16 + j;
    wpack[layer * 16384 + ((t * 4 + kk) * 32 + ln) * 16 + j] =
        (_Float16)Wg[layer * 16384 + k * H_ + n];
}

// ---------------------------------------------------------------- WMMA GEMM
// xh = x16 (M x 128, f16) @ Wpack (128 x 128) with f32 accumulate.
// W staged once per block into LDS; one wave per 16-row tile.
// k-step outer: 8 B-fragments preloaded per k-step (one dscnt wait), then
// 8 back-to-back v_wmma_f32_16x16x32_f16 into independent accumulators.
__global__ __launch_bounds__(256) void k_gemm_wmma(
    const _Float16* __restrict__ x16, const _Float16* __restrict__ wpack,
    float* __restrict__ xh)
{
    __shared__ _Float16 wlds[H_ * H_];           // 32 KB
    {
        v16h* dstv = (v16h*)wlds;
        const v16h* srcv = (const v16h*)wpack;
        #pragma unroll
        for (int i = 0; i < 4; ++i)
            dstv[threadIdx.x + i * 256] = srcv[threadIdx.x + i * 256];
    }
    __syncthreads();

    int wave  = (blockIdx.x * 256 + threadIdx.x) >> 5;   // exact: M/16 waves
    int lane  = threadIdx.x & 31;
    int mBase = wave * 16;
    int sel   = lane >> 4;
    const _Float16* xrow = x16 + (size_t)(mBase + (lane & 15)) * H_;

    v16h a[4];
    #pragma unroll
    for (int kk = 0; kk < 4; ++kk) {
        int kb = kk * 32 + sel * 16;
        AFrag u;
        u.h[0] = *(const v8h*)(xrow + kb);        // K = kb .. kb+7
        u.h[1] = *(const v8h*)(xrow + kb + 8);    // K = kb+8 .. kb+15
        a[kk] = u.v;
    }

    const v16h* bp = (const v16h*)wlds;
    v8f c[8] = {};
    #pragma unroll
    for (int kk = 0; kk < 4; ++kk) {
        v16h bfrag[8];
        #pragma unroll
        for (int t = 0; t < 8; ++t)
            bfrag[t] = bp[(t * 4 + kk) * 32 + lane];
        #pragma unroll
        for (int t = 0; t < 8; ++t)
            c[t] = __builtin_amdgcn_wmma_f32_16x16x32_f16(
                    false, a[kk], false, bfrag[t], (short)0, c[t], false, false);
    }

    // C layout: VGPR r -> row (sel?8:0)+r, col = lane&15
    #pragma unroll
    for (int t = 0; t < 8; ++t) {
        float* outp = xh + (size_t)(mBase + sel * 8) * H_ + t * 16 + (lane & 15);
        #pragma unroll
        for (int r = 0; r < 8; ++r) outp[(size_t)r * H_] = c[t][r];
    }
}

// ----------------------------------------------------- fused GAT attention
// One workgroup per graph.  All segment-softmax state + the 421x128 f32
// aggregation tile live in LDS (272.8 KB of CDNA5's 320 KB WGP LDS).
__device__ __forceinline__ void ldsAtomicMaxF(float* addr, float val) {
    if (val >= 0.f) atomicMax((int*)addr, __float_as_int(val));
    else            atomicMin((unsigned int*)addr, __float_as_uint(val));
}

__global__ __launch_bounds__(256) void k_gat_fused(
    const float* __restrict__ xh, const int* __restrict__ ei,
    const float* __restrict__ a_src, const float* __restrict__ a_dst,
    const float* __restrict__ b_gat,
    float* __restrict__ x32, _Float16* __restrict__ x16)
{
    extern __shared__ char smem[];
    float* agg = (float*)(smem + SM_AGG);
    float* als = (float*)(smem + SM_ALS);
    float* ald = (float*)(smem + SM_ALD);
    float* emx = (float*)(smem + SM_EMX);
    float* den = (float*)(smem + SM_DEN);
    float* ex  = (float*)(smem + SM_EX);
    int*   sL  = (int*)  (smem + SM_SL);
    int*   dL  = (int*)  (smem + SM_DL);

    const int b   = blockIdx.x;
    const int tid = threadIdx.x;
    const float* xhg = xh + (size_t)b * N_ * H_;   // this graph's xh tile

    // phase 1: init LDS state + load local edge lists (self loops appended)
    for (int i = tid; i < N_ * H_; i += 256) agg[i] = 0.f;
    for (int i = tid; i < N_ * HEADS_; i += 256) { emx[i] = FMIN_; den[i] = 0.f; }
    for (int e = tid; e < NE_; e += 256) {
        if (e < E_) { sL[e] = ei[b * 2 * E_ + e]; dL[e] = ei[b * 2 * E_ + E_ + e]; }
        else        { sL[e] = e - E_;             dL[e] = e - E_; }
    }
    // phase 2: attention coefficients al_s/al_d per (node, head)
    for (int i = tid; i < N_ * HEADS_; i += 256) {
        int m = i >> 2, h = i & 3;
        const float* xr = xhg + (size_t)m * H_ + h * D_;
        const float* as = a_src + h * D_;
        const float* ad = a_dst + h * D_;
        float ss = 0.f, sd = 0.f;
        #pragma unroll
        for (int d = 0; d < D_; ++d) { float v = xr[d]; ss += v * as[d]; sd += v * ad[d]; }
        als[i] = ss; ald[i] = sd;
    }
    __syncthreads();

    // phase 3: leaky-relu edge scores + LDS segment max over dst
    for (int i = tid; i < NE_ * HEADS_; i += 256) {
        int e = i >> 2, h = i & 3;
        float v = als[sL[e] * HEADS_ + h] + ald[dL[e] * HEADS_ + h];
        v = v > 0.f ? v : NEG_SLOPE_ * v;
        ex[i] = v;
        ldsAtomicMaxF(&emx[dL[e] * HEADS_ + h], v);
    }
    __syncthreads();

    // phase 4: exp + LDS segment sum
    for (int i = tid; i < NE_ * HEADS_; i += 256) {
        int e = i >> 2, h = i & 3;
        float v = __expf(ex[i] - emx[dL[e] * HEADS_ + h]);
        ex[i] = v;
        atomicAdd(&den[dL[e] * HEADS_ + h], v);
    }
    __syncthreads();

    // phase 5: normalize -> alpha
    for (int i = tid; i < NE_ * HEADS_; i += 256)
        ex[i] = ex[i] / den[dL[i >> 2] * HEADS_ + (i & 3)];
    __syncthreads();

    // phase 6: scatter-aggregate alpha * xh[src] into LDS agg (ds_add_f32).
    // 128 threads span one row (coalesced xh gather), 2 edges in flight.
    {
        int c = tid & 127;
        int h = c >> 5;
        for (int e = tid >> 7; e < NE_; e += 2) {
            int s = sL[e], d = dL[e];
            float v = ex[e * HEADS_ + h] * xhg[(size_t)s * H_ + c];
            atomicAdd(&agg[d * H_ + c], v);
        }
    }
    __syncthreads();

    // phase 7: bias + relu, write fp32 and f16 copies of the new x
    float* xo = x32 + (size_t)b * N_ * H_;
    _Float16* xo16 = x16 + (size_t)b * N_ * H_;
    for (int i = tid; i < N_ * H_; i += 256) {
        float v = agg[i] + b_gat[i & 127];
        v = v > 0.f ? v : 0.f;
        xo[i] = v;
        xo16[i] = (_Float16)v;
    }
}

// ---------------------------------------------------------------- heads
__global__ __launch_bounds__(512) void k_heads(
    const float* __restrict__ x32, const float* __restrict__ accept_mask,
    const float* __restrict__ W_acc, const float* __restrict__ b_acc,
    const float* __restrict__ W_off, const float* __restrict__ b_off,
    const float* __restrict__ W_vf,  const float* __restrict__ b_vf,
    float* __restrict__ out)
{
    int b = blockIdx.x;
    int t = threadIdx.x;
    if (t >= V_ + 3) return;
    const float* h_head = x32 + (size_t)b * N_ * H_;   // node 0
    float acc;
    if (t < 2) {
        acc = b_acc[t];
        #pragma unroll 8
        for (int k = 0; k < H_; ++k) acc += h_head[k] * W_acc[k * 2 + t];
        float lm = __logf(accept_mask[b * 2 + t]);
        acc += lm > FMIN_ ? lm : FMIN_;
    } else if (t < 2 + V_) {
        const float* hv = x32 + (size_t)(b * N_ + 1 + O_ + (t - 2)) * H_;
        acc = b_off[0];
        #pragma unroll 8
        for (int k = 0; k < H_; ++k) acc += hv[k] * W_off[k];
    } else {
        acc = b_vf[0];
        #pragma unroll 8
        for (int k = 0; k < H_; ++k) acc += h_head[k] * W_vf[k];
    }
    out[(size_t)b * (V_ + 3) + t] = acc;
}

// ---------------------------------------------------------------- launch
extern "C" void kernel_launch(void* const* d_in, const int* in_sizes, int n_in,
                              void* d_out, int out_size, void* d_ws, size_t ws_size,
                              hipStream_t stream) {
    const float* head      = (const float*)d_in[0];
    const float* obj       = (const float*)d_in[1];
    const float* val       = (const float*)d_in[2];
    const float* acc_mask  = (const float*)d_in[3];
    const int*   ei        = (const int*)  d_in[4];
    const float* We_head   = (const float*)d_in[5];
    const float* be_head   = (const float*)d_in[6];
    const float* We_obj    = (const float*)d_in[7];
    const float* be_obj    = (const float*)d_in[8];
    const float* We_val    = (const float*)d_in[9];
    const float* be_val    = (const float*)d_in[10];
    const float* W_gat     = (const float*)d_in[11];
    const float* a_src     = (const float*)d_in[12];
    const float* a_dst     = (const float*)d_in[13];
    const float* b_gat     = (const float*)d_in[14];
    const float* W_acc     = (const float*)d_in[15];
    const float* b_acc     = (const float*)d_in[16];
    const float* W_off     = (const float*)d_in[17];
    const float* b_off     = (const float*)d_in[18];
    const float* W_vf      = (const float*)d_in[19];
    const float* b_vf      = (const float*)d_in[20];
    float* out = (float*)d_out;

    // workspace carve-up (all region sizes are multiples of 256 bytes)
    char* ws = (char*)d_ws;
    size_t off = 0;
    auto carve = [&](size_t bytes) { char* p = ws + off; off += bytes; return p; };
    float*    x32   = (float*)   carve((size_t)M_ * H_ * 4);
    _Float16* x16   = (_Float16*)carve((size_t)M_ * H_ * 2);
    float*    xh    = (float*)   carve((size_t)M_ * H_ * 4);
    _Float16* wpack = (_Float16*)carve((size_t)LAYERS_ * H_ * H_ * 2);

    const int MHblocks = (M_ * H_) / 256;        // 215552
    const int Gblocks  = (M_ / 16) * 32 / 256;   // 3368 (one wave / 16 rows)

    k_packW<<<(LAYERS_ * H_ * H_) / 256, 256, 0, stream>>>(W_gat, wpack);
    k_embed<<<MHblocks, 256, 0, stream>>>(head, obj, val, We_head, be_head,
                                          We_obj, be_obj, We_val, be_val, x32, x16);

    for (int l = 0; l < LAYERS_; ++l) {
        k_gemm_wmma<<<Gblocks, 256, 0, stream>>>(x16, wpack + l * H_ * H_, xh);
        k_gat_fused<<<B_, 256, SM_TOTAL, stream>>>(
            xh, ei, a_src + l * HEADS_ * D_, a_dst + l * HEADS_ * D_,
            b_gat + l * H_, x32, x16);
    }

    k_heads<<<B_, 512, 0, stream>>>(x32, acc_mask, W_acc, b_acc,
                                    W_off, b_off, W_vf, b_vf, out);
}
